// NNUE_79843442033240
// MI455X (gfx1250) — compile-verified, matrix operand order checked by reading, not run on hardware
//
#include <hip/hip_runtime.h>
#include <hip/hip_fp16.h>
#include <cmath>

typedef _Float16 v16h __attribute__((ext_vector_type(16)));
typedef float    v8f  __attribute__((ext_vector_type(8)));

#define POS_PER_BLOCK 64
#define THREADS 128   // 4 waves (wave32)

__device__ __forceinline__ int lower_bound_i32(const int* __restrict__ a, int n, int key) {
    int lo = 0, hi = n;
    while (lo < hi) {
        int mid = (lo + hi) >> 1;
        if (a[mid] < key) lo = mid + 1; else hi = mid;
    }
    return lo;
}

__global__ __launch_bounds__(THREADS)
void nnue_fused_kernel(const int*   __restrict__ wfeat, const int* __restrict__ widx,
                       const int*   __restrict__ bfeat, const int* __restrict__ bidx,
                       const float* __restrict__ W1,    const float* __restrict__ b1,
                       const float* __restrict__ W2,    const float* __restrict__ b2,
                       const float* __restrict__ W3,    const float* __restrict__ b3,
                       float* __restrict__ out, int TOT, int NF, int B)
{
    // 56 KB LDS total: [0,32K) f32 acc (phase 1) -> reused as a2 scratch (phase 2/3)
    //                  [32K,48K) f16 a1, [48K,56K) f16 W2
    __shared__ __align__(16) unsigned char smem[56 * 1024];
    float*    acc = (float*)smem;                      // [64][128] f32, 32 KB
    _Float16* a1h = (_Float16*)(smem + 32 * 1024);     // [64][128] f16, 16 KB
    _Float16* w2h = (_Float16*)(smem + 48 * 1024);     // [32][128] f16,  8 KB
    float*    a2s = (float*)smem;                      // [64][32]  f32,  8 KB (reuses acc)

    const int t  = threadIdx.x;
    const int p0 = blockIdx.x * POS_PER_BLOCK;
    const int p1 = min(p0 + POS_PER_BLOCK, B);

    // ---- Phase 0: zero accumulators + stage W2 as f16 in LDS ---------------
    for (int i = t; i < POS_PER_BLOCK * 128; i += THREADS)
        acc[i] = 0.0f;
    for (int i = t; i < 32 * 128; i += THREADS)
        w2h[i] = (_Float16)W2[i];
    __syncthreads();

    // ---- Phase 1: sorted segment-sum gather (no atomics) -------------------
    // threads 0..63 -> white (channels 0..63), threads 64..127 -> black (64..127)
    {
        const bool black = (t >= 64);
        const int  ch    = t & 63;
        const int* feats = black ? bfeat : wfeat;
        const int* idxs  = black ? bidx  : widx;
        const int  cofs  = black ? 64 : 0;

        const int f0 = lower_bound_i32(idxs, TOT, p0);
        const int f1 = lower_bound_i32(idxs, TOT, p1);
        #pragma unroll 4
        for (int f = f0; f < f1; ++f) {
            const int feat = feats[f];          // uniform across the 64-thread group
            const int pos  = idxs[f] - p0;
            // W1T[feat][ch] == W1[ch][feat]  (W1 is [64, NF] row-major, L2-resident)
            acc[pos * 128 + cofs + ch] += W1[ch * NF + feat];
        }
    }
    __syncthreads();

    // ---- bias + hardtanh(0,1) + f16 convert --------------------------------
    for (int i = t; i < POS_PER_BLOCK * 128; i += THREADS) {
        const int c = i & 127;
        float v = acc[i] + b1[c & 63];
        v = fminf(fmaxf(v, 0.0f), 1.0f);        // values in [0,1] -> f16 is safe
        a1h[i] = (_Float16)v;
    }
    __syncthreads();        // acc region is dead past this point -> reused as a2s

    // ---- Phase 2: layer 2 via v_wmma_f32_16x16x32_f16 ----------------------
    // wave w owns positions [w*16, w*16+16); 2 N-tiles x 4 K-chunks of WMMA.
    {
        const int wave  = t >> 5;
        const int lane  = t & 31;
        const int lrow  = lane & 15;
        const int kbase = (lane < 16) ? 0 : 8;   // ISA 16-bit A/B-matrix lane split

        for (int n = 0; n < 2; ++n) {
            v8f c;
            const float bias = b2[n * 16 + lrow];     // C/D: N = lane%16
            #pragma unroll
            for (int r = 0; r < 8; ++r) c[r] = bias;  // bias folded into C

            #pragma unroll
            for (int k = 0; k < 4; ++k) {
                v16h a, bm;
                const int m    = wave * 16 + lrow;    // A: M = lane%16
                const int ncol = n    * 16 + lrow;    // B: N = lane%16
                #pragma unroll
                for (int h = 0; h < 16; ++h) {
                    // half h -> K per ISA layout: VGPR0-3 = base+0..7, VGPR4-7 = base+16..23
                    // contiguous 8-half (16 B) runs -> ds_load_b128
                    const int kk = k * 32 + kbase + ((h < 8) ? h : (8 + h));
                    a[h]  = a1h[m * 128 + kk];
                    bm[h] = w2h[ncol * 128 + kk];     // W2.T[kk][ncol], f16-staged in LDS
                }
                // D = A x B + C   (8 args: neg_a, A, neg_b, B, c_mod, C, reuse_a, reuse_b)
                c = __builtin_amdgcn_wmma_f32_16x16x32_f16(
                        false, a, false, bm, (short)0, c, false, false);
            }

            // D layout: lanes 0-15 -> M=r, lanes 16-31 -> M=r+8; N = lane%16
            #pragma unroll
            for (int r = 0; r < 8; ++r) {
                const int m = wave * 16 + r + ((lane < 16) ? 0 : 8);
                a2s[m * 32 + n * 16 + lrow] = fminf(fmaxf(c[r], 0.0f), 1.0f);
            }
        }
    }
    __syncthreads();

    // ---- Phase 3: 32->1 dot + sigmoid --------------------------------------
    if (t < POS_PER_BLOCK && (p0 + t) < B) {
        float s = b3[0];
        #pragma unroll
        for (int nn = 0; nn < 32; ++nn)
            s += a2s[t * 32 + nn] * W3[nn];
        out[p0 + t] = 1.0f / (1.0f + __expf(-s));
    }
}

extern "C" void kernel_launch(void* const* d_in, const int* in_sizes, int n_in,
                              void* d_out, int out_size, void* d_ws, size_t ws_size,
                              hipStream_t stream) {
    const int*   wfeat = (const int*)  d_in[0];
    const int*   widx  = (const int*)  d_in[1];
    const int*   bfeat = (const int*)  d_in[2];
    const int*   bidx  = (const int*)  d_in[3];
    const float* W1    = (const float*)d_in[4];
    const float* b1    = (const float*)d_in[5];
    const float* W2    = (const float*)d_in[6];
    const float* b2    = (const float*)d_in[7];
    const float* W3    = (const float*)d_in[8];
    const float* b3    = (const float*)d_in[9];
    float*       out   = (float*)d_out;

    const int TOT = in_sizes[0];
    const int NF  = in_sizes[4] / 64;   // W1 is [64, NF]
    const int B   = out_size;           // one output per position

    const int blocks = (B + POS_PER_BLOCK - 1) / POS_PER_BLOCK;
    nnue_fused_kernel<<<blocks, THREADS, 0, stream>>>(
        wfeat, widx, bfeat, bidx, W1, b1, W2, b2, W3, b3, out, TOT, NF, B);
}
